// AttentionBlock_28793460752829
// MI455X (gfx1250) — matrix operand order repaired and checked by reading, not compiled, not run on hardware
//
#include <hip/hip_runtime.h>
#include <hip/hip_fp16.h>

// ---------------------------------------------------------------------------
// Problem constants (reference: B=16, S=8, D=2048, H=16, HD=128, MAXSEQ=4096)
// ---------------------------------------------------------------------------
#define B_      16
#define S_      8
#define D_      2048
#define H_      16
#define HD_     128
#define MAXSEQ_ 4096
#define RROWS_  (B_ * S_)      // 128 token rows
#define KVSTR_  (H_ * HD_)     // 2048 floats between consecutive cache rows
#define SOFTCAP_ 50.0f
#define EPS_     1e-6f
#define QK_SCALE_ 0.08838834764831845f   // 1/sqrt(128)

typedef __attribute__((ext_vector_type(16))) _Float16 v16h;
typedef __attribute__((ext_vector_type(8)))  float    v8f;
typedef __attribute__((ext_vector_type(4)))  unsigned int u32x4;
typedef __attribute__((ext_vector_type(8)))  int      i32x8;
typedef __attribute__((ext_vector_type(4)))  int      i32x4;

#if __has_builtin(__builtin_amdgcn_tensor_load_to_lds) && \
    __has_builtin(__builtin_amdgcn_tensor_store_from_lds) && \
    __has_builtin(__builtin_amdgcn_s_wait_tensorcnt)
#define HAVE_TDM 1
#else
#define HAVE_TDM 0
#endif

// ---------------------------------------------------------------------------
// Kernel 1: streaming cache copy (the dominant, HBM-bound part: 2 x 512 MB).
// TDM path: one wave per 64KB chunk, global -> LDS -> global via the Tensor
// Data Mover; tracked with TENSORcnt. Descriptor packing per
// CDNA5 ISA (D# group 0/1; 1-D tile, 8B elements).
// ---------------------------------------------------------------------------
#define TDM_CHUNK_BYTES  65536u
#define TDM_CHUNK_E8     (TDM_CHUNK_BYTES / 8u)   // 8192 8-byte elements

#if HAVE_TDM
__device__ __forceinline__ u32x4 tdm_group0(unsigned lds_off, unsigned long long ga) {
    u32x4 g;
    g[0] = 1u;                                   // count=1, is_restore=0, gather off
    g[1] = lds_off;                              // lds_addr (bytes)
    g[2] = (unsigned)ga;                         // global_addr[31:0]
    g[3] = ((unsigned)(ga >> 32) & 0x01FFFFFFu)  // global_addr[56:32]
           | 0x80000000u;                        // type = 2 ("image"), bits 127:126
    return g;
}

__device__ __forceinline__ i32x8 tdm_group1(unsigned n_e8) {
    i32x8 g;
    g[0] = 0x00030000;                           // data_size = 3 (8 bytes); mask/flags 0
    g[1] = (int)((n_e8 & 0xFFFFu) << 16);        // tensor_dim0[15:0]
    g[2] = (int)((n_e8 >> 16) | (1u << 16));     // tensor_dim0[31:16] | tensor_dim1=1
    g[3] = (int)((n_e8 & 0xFFFFu) << 16);        // tensor_dim1 hi=0 | tile_dim0 = n_e8
    g[4] = 0;                                    // tile_dim1 = 0 (unused), tile_dim2 = 0
    g[5] = (int)n_e8;                            // tensor_dim0_stride[31:0]
    g[6] = 0;                                    // stride0 hi | stride1 lo
    g[7] = 0;                                    // stride1 hi
    return g;
}
#endif

__global__ void cache_copy_kernel(const void* __restrict__ src,
                                  void* __restrict__ dst,
                                  unsigned long long nbytes) {
#if HAVE_TDM
    __shared__ unsigned long long lds_buf[TDM_CHUNK_E8];
    // Generic->int truncation of a flat LDS pointer yields the LDS byte offset
    // (flat LDS aperture: low 32 bits are the offset). Keeps LDS allocated.
    const unsigned lds_off = (unsigned)(unsigned long long)(void*)&lds_buf[0];

    const unsigned long long nchunks = nbytes / TDM_CHUNK_BYTES;
    const i32x4 zero4 = {0, 0, 0, 0};
    const i32x8 zero8 = {0, 0, 0, 0, 0, 0, 0, 0};
    const i32x8 g1 = tdm_group1(TDM_CHUNK_E8);

    for (unsigned long long c = blockIdx.x; c < nchunks; c += gridDim.x) {
        const unsigned long long soff = c * (unsigned long long)TDM_CHUNK_BYTES;
        // DMA: global -> LDS
        u32x4 g0l = tdm_group0(lds_off, (unsigned long long)src + soff);
        __builtin_amdgcn_tensor_load_to_lds(g0l, g1, zero4, zero4, zero8, 0);
        __builtin_amdgcn_s_wait_tensorcnt(0);
        // DMA: LDS -> global
        u32x4 g0s = tdm_group0(lds_off, (unsigned long long)dst + soff);
        __builtin_amdgcn_tensor_store_from_lds(g0s, g1, zero4, zero4, zero8, 0);
        __builtin_amdgcn_s_wait_tensorcnt(0);
    }
#else
    // Fallback: plain wide vector copy (also what the host stub pass sees).
    const float4* s = (const float4*)src;
    float4* d = (float4*)dst;
    size_t n4 = (size_t)(nbytes / 16u);
    size_t i = (size_t)blockIdx.x * blockDim.x + threadIdx.x;
    size_t stride = (size_t)gridDim.x * blockDim.x;
    for (; i < n4; i += stride) d[i] = s[i];
#endif
}

// ---------------------------------------------------------------------------
// Kernel 2: generic WMMA GEMM  C[M,N] = A[M,K] * W[K,N] + bias[N]
// A, W are f32 row-major; converted to f16 in-register; f32 accumulate.
// One wave (32 threads) computes one 16x16 tile via v_wmma_f32_16x16x32_f16.
// Fragment layouts used here (contiguous k, same in A and B):
//   A (16x32 f16): lane half g (0/1): elems j=0..7  -> K = g*16 + j
//                                     elems j=8..15 -> K = g*16 + j
//   B (32x16 f16): lane half g: elem j=0..15 -> K = g*16 + j (as A, not ISA table order), col = lane%16
//   C/D (16x16 f32): vgpr r -> row = r + 8*g, col = lane%16
// ---------------------------------------------------------------------------
__device__ __forceinline__ v8f wmma_step(const float* __restrict__ arow,
                                         const float* __restrict__ W,
                                         int k0, int half_g, int col, int N, v8f acc) {
    v16h a, b;
    const int abase = k0 + half_g * 16;
#pragma unroll
    for (int j = 0; j < 8; ++j) {
        a[j]     = (_Float16)arow[abase + j];
        a[j + 8] = (_Float16)arow[abase + 8 + j];
    }
    const int bbase = k0 + half_g * 16;
#pragma unroll
    for (int j = 0; j < 16; ++j) {
        b[j] = (_Float16)W[(size_t)(bbase + j) * N + col];
    }
    return __builtin_amdgcn_wmma_f32_16x16x32_f16(
        /*neg_a=*/false, a, /*neg_b=*/false, b,
        /*c_mod=*/(short)0, acc, /*reuse_a=*/false, /*reuse_b=*/false);
}

__global__ void gemm_wmma_kernel(const float* __restrict__ A,
                                 const float* __restrict__ W,
                                 const float* __restrict__ bias,
                                 float* __restrict__ C,
                                 int M, int N, int K) {
    const int tile_n = blockIdx.x;
    const int tile_m = blockIdx.y;
    const int lane   = threadIdx.x & 31;
    const int half_g = lane >> 4;      // 0 or 1
    const int mn     = lane & 15;
    const int row = tile_m * 16 + mn;  // A row this lane reads
    const int col = tile_n * 16 + mn;  // C col this lane owns

    const float* __restrict__ arow = A + (size_t)row * K;

    v8f acc = {0.f, 0.f, 0.f, 0.f, 0.f, 0.f, 0.f, 0.f};

    for (int k0 = 0; k0 < K; k0 += 64) {
        acc = wmma_step(arow, W, k0,      half_g, col, N, acc);
        acc = wmma_step(arow, W, k0 + 32, half_g, col, N, acc);
    }

    const float bval = bias ? bias[col] : 0.f;
#pragma unroll
    for (int r = 0; r < 8; ++r) {
        const int orow = tile_m * 16 + half_g * 8 + r;
        C[(size_t)orow * N + col] = acc[r] + bval;
    }
}

// ---------------------------------------------------------------------------
// Kernel 3: finalize — RMSNorm + RoPE (+ 1/sqrt(HD) for q), KV scatter.
// One wave per 128-element row; lane L owns elems {L, L+32, L+64, L+96}.
// ---------------------------------------------------------------------------
__device__ __forceinline__ float wave_sum32(float v) {
#pragma unroll
    for (int off = 16; off >= 1; off >>= 1) v += __shfl_xor(v, off, 32);
    return v;
}

__global__ void finalize_kernel(float* __restrict__ q_buf,
                                const float* __restrict__ k_buf,
                                const float* __restrict__ v_buf,
                                const int* __restrict__ seq_pos,
                                const float* __restrict__ q_scale,
                                const float* __restrict__ k_scale,
                                float* __restrict__ key_all,
                                float* __restrict__ val_all) {
    const int bid  = blockIdx.x;
    const int lane = threadIdx.x;   // 0..31

    if (bid < RROWS_) {
        // ---- q: rmsnorm + rope + qk-scale, in place ----
        const int r = bid;                    // r = b*S + s
        const int b = r >> 3, s = r & 7;
        const int pos = seq_pos[b * S_ + s];
        float* p = q_buf + (size_t)r * HD_;

        float x0 = p[lane], x1 = p[lane + 32], x2 = p[lane + 64], x3 = p[lane + 96];
        float ss = wave_sum32(x0 * x0 + x1 * x1 + x2 * x2 + x3 * x3);
        float inv = rsqrtf(ss * (1.0f / HD_) + EPS_);
        x0 *= inv * q_scale[lane];
        x1 *= inv * q_scale[lane + 32];
        x2 *= inv * q_scale[lane + 64];
        x3 *= inv * q_scale[lane + 96];
        float ia = (float)lane, ib = (float)(lane + 32);
        float fp = (float)pos;
        float sa = fp * powf(10000.0f, -ia * (1.0f / 64.0f));
        float sb = fp * powf(10000.0f, -ib * (1.0f / 64.0f));
        float sna = sinf(sa), csa = cosf(sa);
        float snb = sinf(sb), csb = cosf(sb);
        float o0 = x0 * csa - x2 * sna, o2 = x2 * csa + x0 * sna;
        float o1 = x1 * csb - x3 * snb, o3 = x3 * csb + x1 * snb;
        p[lane]      = o0 * QK_SCALE_;
        p[lane + 32] = o1 * QK_SCALE_;
        p[lane + 64] = o2 * QK_SCALE_;
        p[lane + 96] = o3 * QK_SCALE_;
    } else if (bid < RROWS_ + RROWS_ * H_) {
        // ---- k: rmsnorm + rope, scatter into key_all ----
        const int idx = bid - RROWS_;         // (b*S+s)*H + h
        const int r = idx >> 4, h = idx & 15;
        const int b = r >> 3, s = r & 7;
        const int pos = seq_pos[b * S_ + s];
        const float* p = k_buf + (size_t)r * KVSTR_ + h * HD_;

        float x0 = p[lane], x1 = p[lane + 32], x2 = p[lane + 64], x3 = p[lane + 96];
        float ss = wave_sum32(x0 * x0 + x1 * x1 + x2 * x2 + x3 * x3);
        float inv = rsqrtf(ss * (1.0f / HD_) + EPS_);
        x0 *= inv * k_scale[lane];
        x1 *= inv * k_scale[lane + 32];
        x2 *= inv * k_scale[lane + 64];
        x3 *= inv * k_scale[lane + 96];
        float ia = (float)lane, ib = (float)(lane + 32);
        float fp = (float)pos;
        float sa = fp * powf(10000.0f, -ia * (1.0f / 64.0f));
        float sb = fp * powf(10000.0f, -ib * (1.0f / 64.0f));
        float sna = sinf(sa), csa = cosf(sa);
        float snb = sinf(sb), csb = cosf(sb);
        float o0 = x0 * csa - x2 * sna, o2 = x2 * csa + x0 * sna;
        float o1 = x1 * csb - x3 * snb, o3 = x3 * csb + x1 * snb;

        float* d = key_all + (((size_t)b * MAXSEQ_ + pos) * H_ + h) * HD_;
        d[lane] = o0; d[lane + 32] = o1; d[lane + 64] = o2; d[lane + 96] = o3;
    } else {
        // ---- v: raw scatter into val_all ----
        const int idx = bid - RROWS_ - RROWS_ * H_;
        const int r = idx >> 4, h = idx & 15;
        const int b = r >> 3, s = r & 7;
        const int pos = seq_pos[b * S_ + s];
        const float* p = v_buf + (size_t)r * KVSTR_ + h * HD_;
        float* d = val_all + (((size_t)b * MAXSEQ_ + pos) * H_ + h) * HD_;
        d[lane]      = p[lane];
        d[lane + 32] = p[lane + 32];
        d[lane + 64] = p[lane + 64];
        d[lane + 96] = p[lane + 96];
    }
}

// ---------------------------------------------------------------------------
// Kernel 4: WMMA flash attention — one wave per (b,h) handles all 8 queries
// (padded to M=16; A rows 8..15 are zero). Per 32-key chunk:
//   - 2x QK tiles (4 WMMAs each, K-dim = HD = 128)
//   - single online-softmax update over 32 keys (mask -> tanh softcap -> exp),
//     masked keys clamped to exactly -50 (== reference tanh(big_neg/50)*50)
//   - P transposed through a 2KB LDS tile (per-wave DS ops are in-order)
//   - 8x PV column-tile WMMAs with accumulator rescale
// Tail keys beyond the last chunk enter the denominator exactly as -50 terms.
// ---------------------------------------------------------------------------
__global__ __launch_bounds__(32)
void attention_wmma_kernel(const float* __restrict__ q_buf,
                           const float* __restrict__ key_all,
                           const float* __restrict__ val_all,
                           const int* __restrict__ seq_pos,
                           float* __restrict__ x_buf) {
    __shared__ float p_lds[16][32];         // [query row][key within chunk]

    const int bid  = blockIdx.x;            // b*H + h
    const int b    = bid >> 4;
    const int h    = bid & 15;
    const int lane = threadIdx.x & 31;
    const int half_g = lane >> 4;           // 0 or 1
    const int mn     = lane & 15;

    int pos[8];
#pragma unroll
    for (int r = 0; r < 8; ++r) pos[r] = seq_pos[b * S_ + r];
    const int pos_max = pos[7];             // seq_pos increases with s

    // ---- Q A-fragments (constant across chunks); rows 8..15 zeroed ----
    v16h qf[4];
    {
        const float* qrow = q_buf + (size_t)(b * S_ + (mn & 7)) * HD_;
        const bool valid = (mn < 8);
#pragma unroll
        for (int t = 0; t < 4; ++t) {
            const int ch = 32 * t + half_g * 16;
#pragma unroll
            for (int j = 0; j < 8; ++j) {
                qf[t][j]     = valid ? (_Float16)qrow[ch + j]      : (_Float16)0.f;
                qf[t][j + 8] = valid ? (_Float16)qrow[ch + 8 + j]  : (_Float16)0.f;
            }
        }
    }

    const float* Kb = key_all + (size_t)b * MAXSEQ_ * KVSTR_ + (size_t)h * HD_;
    const float* Vb = val_all + (size_t)b * MAXSEQ_ * KVSTR_ + (size_t)h * HD_;

    float m_r[8], l_r[8];
#pragma unroll
    for (int r = 0; r < 8; ++r) { m_r[r] = -1e30f; l_r[r] = 0.f; }

    v8f o_acc[8];
#pragma unroll
    for (int t = 0; t < 8; ++t) o_acc[t] = (v8f){0,0,0,0,0,0,0,0};

    const int nchunks = (pos_max >> 5) + 1;  // 32 keys per chunk

    for (int c = 0; c < nchunks; ++c) {
        const int kb = c * 32;

        if (c + 1 < nchunks) {  // uniform condition: prefetch next chunk's rows
            __builtin_prefetch(Kb + (size_t)(kb + 32 + mn) * KVSTR_, 0, 3);
            __builtin_prefetch(Vb + (size_t)(kb + 32 + mn) * KVSTR_, 0, 3);
        }

        // ---- QK: two 16-key logit tiles ----
        v8f lg0 = (v8f){0,0,0,0,0,0,0,0};
        v8f lg1 = (v8f){0,0,0,0,0,0,0,0};
        {
            const float* kr0 = Kb + (size_t)(kb + mn) * KVSTR_;
            const float* kr1 = Kb + (size_t)(kb + 16 + mn) * KVSTR_;
#pragma unroll
            for (int t = 0; t < 4; ++t) {
                const int ch0 = 32 * t + half_g * 16;
                v16h bf0, bf1;
#pragma unroll
                for (int j = 0; j < 16; ++j) {
                    bf0[j] = (_Float16)kr0[ch0 + j];
                    bf1[j] = (_Float16)kr1[ch0 + j];
                }
                lg0 = __builtin_amdgcn_wmma_f32_16x16x32_f16(false, qf[t], false, bf0,
                                                             (short)0, lg0, false, false);
                lg1 = __builtin_amdgcn_wmma_f32_16x16x32_f16(false, qf[t], false, bf1,
                                                             (short)0, lg1, false, false);
            }
        }

        // ---- online softmax over all 32 keys of this chunk ----
        float scale_r[8];
        const int key0 = kb + mn;
        const int key1 = kb + 16 + mn;
#pragma unroll
        for (int r = 0; r < 8; ++r) {
            float v0 = (key0 <= pos[r])
                       ? tanhf(lg0[r] * (1.0f / SOFTCAP_)) * SOFTCAP_ : -SOFTCAP_;
            float v1 = (key1 <= pos[r])
                       ? tanhf(lg1[r] * (1.0f / SOFTCAP_)) * SOFTCAP_ : -SOFTCAP_;
            float cm = fmaxf(v0, v1);
            cm = fmaxf(cm, __shfl_xor(cm, 1, 32));
            cm = fmaxf(cm, __shfl_xor(cm, 2, 32));
            cm = fmaxf(cm, __shfl_xor(cm, 4, 32));
            cm = fmaxf(cm, __shfl_xor(cm, 8, 32));
            const float mnew = fmaxf(m_r[r], cm);
            const float sc = __expf(m_r[r] - mnew);
            const float p0 = __expf(v0 - mnew);
            const float p1 = __expf(v1 - mnew);
            float ps = p0 + p1;
            ps += __shfl_xor(ps, 1, 32);
            ps += __shfl_xor(ps, 2, 32);
            ps += __shfl_xor(ps, 4, 32);
            ps += __shfl_xor(ps, 8, 32);
            l_r[r] = l_r[r] * sc + ps;
            m_r[r] = mnew;
            scale_r[r] = sc;
            // stage P (transposed) through LDS; padded rows 8..15 get zeros
            const int row = r + 8 * half_g;
            p_lds[row][mn]      = half_g ? 0.f : p0;
            p_lds[row][16 + mn] = half_g ? 0.f : p1;
        }

        // ---- P A-fragment (16x32, M=query, K=key) from LDS ----
        v16h pf;
#pragma unroll
        for (int j = 0; j < 8; ++j) {
            pf[j]     = (_Float16)p_lds[mn][half_g * 16 + j];
            pf[j + 8] = (_Float16)p_lds[mn][half_g * 16 + 8 + j];
        }

        // ---- PV: 8 column tiles of HD=128, accumulator rescale ----
#pragma unroll
        for (int t = 0; t < 8; ++t) {
            v16h vf;
#pragma unroll
            for (int j = 0; j < 16; ++j) {
                vf[j] = (_Float16)Vb[(size_t)(kb + half_g * 16 + j) * KVSTR_ + t * 16 + mn];
            }
            v8f o = o_acc[t];
#pragma unroll
            for (int r = 0; r < 8; ++r) o[r] *= scale_r[r];
            o_acc[t] = __builtin_amdgcn_wmma_f32_16x16x32_f16(false, pf, false, vf,
                                                              (short)0, o, false, false);
        }
    }

    // ---- exact -50 tail for keys past the processed range ----
#pragma unroll
    for (int r = 0; r < 8; ++r)
        l_r[r] += (float)(MAXSEQ_ - 1 - pos[r]) * __expf(-SOFTCAP_ - m_r[r]);

    // ---- store: rows 0..7 live in half-group 0 lanes ----
    if (half_g == 0) {
#pragma unroll
        for (int r = 0; r < 8; ++r) {
            const float invl = 1.0f / l_r[r];
            float* xr = x_buf + (size_t)(b * S_ + r) * KVSTR_ + h * HD_;
#pragma unroll
            for (int t = 0; t < 8; ++t)
                xr[t * 16 + mn] = o_acc[t][r] * invl;
        }
    }
}

// ---------------------------------------------------------------------------
// Host launcher
// ---------------------------------------------------------------------------
extern "C" void kernel_launch(void* const* d_in, const int* in_sizes, int n_in,
                              void* d_out, int out_size, void* d_ws, size_t ws_size,
                              hipStream_t stream) {
    const float* inputs   = (const float*)d_in[0];
    const int*   seq_pos  = (const int*)  d_in[1];
    const float* cache_k  = (const float*)d_in[2];
    const float* cache_v  = (const float*)d_in[3];
    const float* wq = (const float*)d_in[4];
    const float* bq = (const float*)d_in[5];
    const float* wk = (const float*)d_in[6];
    const float* bk = (const float*)d_in[7];
    const float* wv = (const float*)d_in[8];
    const float* bv = (const float*)d_in[9];
    const float* wo = (const float*)d_in[10];
    const float* bo = (const float*)d_in[11];
    const float* q_scale = (const float*)d_in[12];
    const float* k_scale = (const float*)d_in[13];

    float* out = (float*)d_out;                                  // [B,S,D]
    float* key_all = out + (size_t)RROWS_ * D_;                  // [B,MAXSEQ,H,HD]
    float* val_all = key_all + (size_t)B_ * MAXSEQ_ * H_ * HD_;  // [B,MAXSEQ,H,HD]

    float* ws    = (float*)d_ws;
    float* q_buf = ws;                                   // [128][128]
    float* k_buf = q_buf + (size_t)RROWS_ * HD_;         // [128][2048]
    float* v_buf = k_buf + (size_t)RROWS_ * KVSTR_;      // [128][2048]
    float* x_buf = v_buf + (size_t)RROWS_ * KVSTR_;      // [128][2048]

    const unsigned long long cache_bytes =
        (unsigned long long)B_ * MAXSEQ_ * H_ * HD_ * sizeof(float);  // 512MB
    const unsigned nchunks = (unsigned)(cache_bytes / TDM_CHUNK_BYTES);  // 8192

    // 1) TDM-stream old caches into the output buffers (the HBM-bound bulk)
    cache_copy_kernel<<<nchunks, 32, 0, stream>>>(cache_k, key_all, cache_bytes);
    cache_copy_kernel<<<nchunks, 32, 0, stream>>>(cache_v, val_all, cache_bytes);

    // 2) WMMA projections: q [128x128], k [128x2048], v [128x2048], K=2048
    gemm_wmma_kernel<<<dim3(HD_ / 16, RROWS_ / 16), 32, 0, stream>>>(
        inputs, wq, bq, q_buf, RROWS_, HD_, D_);
    gemm_wmma_kernel<<<dim3(KVSTR_ / 16, RROWS_ / 16), 32, 0, stream>>>(
        inputs, wk, bk, k_buf, RROWS_, KVSTR_, D_);
    gemm_wmma_kernel<<<dim3(KVSTR_ / 16, RROWS_ / 16), 32, 0, stream>>>(
        inputs, wv, bv, v_buf, RROWS_, KVSTR_, D_);

    // 3) rmsnorm + rope + scatter (after copy so scatters overwrite)
    finalize_kernel<<<RROWS_ + 2 * RROWS_ * H_, 32, 0, stream>>>(
        q_buf, k_buf, v_buf, seq_pos, q_scale, k_scale, key_all, val_all);

    // 4) WMMA flash attention over the updated caches (one wave per (b,h))
    attention_wmma_kernel<<<B_ * H_, 32, 0, stream>>>(
        q_buf, key_all, val_all, seq_pos, x_buf);

    // 5) WMMA output projection: out = x [128x2048] * wo [2048x2048] + bo
    gemm_wmma_kernel<<<dim3(D_ / 16, RROWS_ / 16), 32, 0, stream>>>(
        x_buf, wo, bo, out, RROWS_, D_, KVSTR_);
}
